// Transformer_PostLN_24026047054588
// MI455X (gfx1250) — compile-verified, matrix operand order checked by reading, not run on hardware
//
#include <hip/hip_runtime.h>

// ---------------------------------------------------------------------------
// Types
// ---------------------------------------------------------------------------
typedef unsigned short u16;
typedef __attribute__((ext_vector_type(8)))  __bf16 v8bf;
typedef __attribute__((ext_vector_type(16))) __bf16 v16bf;
typedef __attribute__((ext_vector_type(8)))  float  v8f;

#define WMMA_BF16(a, b, c) \
  __builtin_amdgcn_wmma_f32_16x16x32_bf16(false, (a), false, (b), (short)0, (c), false, false)

// Constants of this problem
#define BB   4
#define LL   2048
#define FEA  512
#define HH   8
#define DKD  64
#define FF   2048
#define MTOT (BB * LL)   // 8192

// ---------------------------------------------------------------------------
// Helpers
// ---------------------------------------------------------------------------
__device__ __forceinline__ u16 f2bf(float f) {
  union { float f; unsigned u; } x; x.f = f;
  unsigned u = x.u;
  u += 0x7FFFu + ((u >> 16) & 1u);   // round-to-nearest-even
  return (u16)(u >> 16);
}

// Load a 16x32 bf16 fragment (A layout; also B layout when the "row" is the
// output column of a [N,K]-stored weight).  Per ISA 7.12.2:
//   lane<16 : K chunks at {0..7} and {16..23};  lane>=16: {8..15} and {24..31}
__device__ __forceinline__ v16bf load_frag(const u16* base, int stride, int lane) {
  int m = lane & 15, half = lane >> 4;
  const u16* p = base + (size_t)m * stride + half * 8;
  union { v16bf v; v8bf h[2]; } f;
  f.h[0] = *(const v8bf*)(p);
  f.h[1] = *(const v8bf*)(p + 16);
  return f.v;
}

// ---------------------------------------------------------------------------
// Conversion kernels
// ---------------------------------------------------------------------------
__global__ void cvt_bf16_kernel(const float* __restrict__ x, u16* __restrict__ y, int n) {
  int i = blockIdx.x * blockDim.x + threadIdx.x;
  if (i < n) y[i] = f2bf(x[i]);
}

// src [K][N] fp32 (row-major, [in,out])  ->  dst [N][K] bf16
__global__ void cvt_tr_bf16_kernel(const float* __restrict__ src, u16* __restrict__ dst,
                                   int K, int N) {
  int i = blockIdx.x * blockDim.x + threadIdx.x;
  if (i < K * N) {
    int k = i / N, n = i % N;
    dst[(size_t)n * K + k] = f2bf(src[i]);
  }
}

// ---------------------------------------------------------------------------
// Generic bf16 WMMA GEMM:  out = A[M,K] @ Wt[N,K]^T + bias
// Block = 256 threads = 8 waves (2x4), wave computes 32x32, block tile 64x128.
// EPI: 0 = f32 store, 1 = bf16 store, 2 = bf16 store transposed as V^T[b,h,d,l],
//      3 = exact GELU -> bf16 store
// ---------------------------------------------------------------------------
template <int EPI>
__global__ __launch_bounds__(256) void gemm_bf16_kernel(
    const u16* __restrict__ A, const u16* __restrict__ Wt,
    const float* __restrict__ bias, float* __restrict__ outF,
    u16* __restrict__ outB, int M, int N, int K) {
  int lane = threadIdx.x & 31, wid = threadIdx.x >> 5;
  int wm = wid >> 2;          // 0..1
  int wn = wid & 3;           // 0..3
  int m0 = blockIdx.x * 64 + wm * 32;
  int n0 = blockIdx.y * 128 + wn * 32;

  v8f acc00 = {}, acc01 = {}, acc10 = {}, acc11 = {};
  const u16* Am0 = A + (size_t)m0 * K;
  const u16* Am1 = A + (size_t)(m0 + 16) * K;
  const u16* Wn0 = Wt + (size_t)n0 * K;
  const u16* Wn1 = Wt + (size_t)(n0 + 16) * K;

  for (int k0 = 0; k0 < K; k0 += 32) {
    if (k0 + 32 < K) {  // global_prefetch_b8 of next K tile
      __builtin_prefetch(Am0 + k0 + 32 + (size_t)(lane & 15) * K, 0, 1);
      __builtin_prefetch(Wn0 + k0 + 32 + (size_t)(lane & 15) * K, 0, 1);
    }
    v16bf a0 = load_frag(Am0 + k0, K, lane);
    v16bf a1 = load_frag(Am1 + k0, K, lane);
    v16bf b0 = load_frag(Wn0 + k0, K, lane);
    v16bf b1 = load_frag(Wn1 + k0, K, lane);
    acc00 = WMMA_BF16(a0, b0, acc00);
    acc01 = WMMA_BF16(a0, b1, acc01);
    acc10 = WMMA_BF16(a1, b0, acc10);
    acc11 = WMMA_BF16(a1, b1, acc11);
  }

  int half = lane >> 4, ln = lane & 15;
#pragma unroll
  for (int v = 0; v < 8; v++) {
    int rb = v + 8 * half;
    float vals[4] = {acc00[v], acc01[v], acc10[v], acc11[v]};
    int rows[4] = {m0 + rb, m0 + rb, m0 + 16 + rb, m0 + 16 + rb};
    int cols[4] = {n0 + ln, n0 + 16 + ln, n0 + ln, n0 + 16 + ln};
#pragma unroll
    for (int t = 0; t < 4; t++) {
      int row = rows[t], col = cols[t];
      float val = vals[t] + bias[col];
      if constexpr (EPI == 0) {
        outF[(size_t)row * N + col] = val;
      } else if constexpr (EPI == 1) {
        outB[(size_t)row * N + col] = f2bf(val);
      } else if constexpr (EPI == 2) {
        int bb = row >> 11, l = row & (LL - 1);
        int hh = col >> 6, d = col & (DKD - 1);
        outB[(((size_t)bb * HH + hh) * DKD + d) * LL + l] = f2bf(val);
      } else {  // exact GELU
        float g = 0.5f * val * (1.0f + erff(val * 0.70710678118654752f));
        outB[(size_t)row * N + col] = f2bf(g);
      }
    }
  }
}

// ---------------------------------------------------------------------------
// Flash attention: each wave handles one 16-row q tile of one (b,h),
// online softmax over key tiles of 32, z accumulated via WMMA with V^T.
// Block = 128 threads = 4 waves.
// ---------------------------------------------------------------------------
__global__ __launch_bounds__(128) void attn_kernel(
    const u16* __restrict__ Qp, const u16* __restrict__ Kp,
    const u16* __restrict__ Vt, const int* __restrict__ maskPAD,
    u16* __restrict__ Zb) {
  __shared__ __align__(16) u16 lds[4 * 16 * 32];

  int lane = threadIdx.x & 31, wid = threadIdx.x >> 5;
  int gw = blockIdx.x * 4 + wid;          // 0..4095
  int bh = gw >> 7;                       // 128 q-tiles per (b,h)
  int qt = gw & 127;
  int b = bh >> 3, h = bh & 7;
  int q0 = qt * 16;
  int half = lane >> 4, ln = lane & 15;

  const u16* Qb = Qp + ((size_t)b * LL) * FEA + h * DKD;  // row stride FEA
  const u16* Kb = Kp + ((size_t)b * LL) * FEA + h * DKD;
  const u16* Vb = Vt + ((size_t)(b * HH + h)) * DKD * LL; // [64][L]
  const int* mb = maskPAD + (size_t)b * LL * LL;
  u16* lw = lds + wid * 16 * 32;

  v16bf qf0 = load_frag(Qb + (size_t)q0 * FEA, FEA, lane);
  v16bf qf1 = load_frag(Qb + (size_t)q0 * FEA + 32, FEA, lane);

  v8f z0 = {}, z1 = {}, z2 = {}, z3 = {};
  float rm[8], rl[8];
#pragma unroll
  for (int v = 0; v < 8; v++) { rm[v] = -3.0e38f; rl[v] = 0.0f; }

  for (int kt = 0; kt < LL; kt += 32) {
    // ---- scores: 16q x 32k, K-dim = DK = 64 (two WMMAs per key subtile) ----
    v16bf kf0a = load_frag(Kb + (size_t)kt * FEA, FEA, lane);
    v16bf kf0b = load_frag(Kb + (size_t)kt * FEA + 32, FEA, lane);
    v16bf kf1a = load_frag(Kb + (size_t)(kt + 16) * FEA, FEA, lane);
    v16bf kf1b = load_frag(Kb + (size_t)(kt + 16) * FEA + 32, FEA, lane);
    v8f s0 = {}, s1 = {};
    s0 = WMMA_BF16(qf0, kf0a, s0);
    s0 = WMMA_BF16(qf1, kf0b, s0);
    s1 = WMMA_BF16(qf0, kf1a, s1);
    s1 = WMMA_BF16(qf1, kf1b, s1);

    // ---- online softmax per row (rows live across lanes in C layout) ----
#pragma unroll
    for (int v = 0; v < 8; v++) {
      int q = q0 + v + 8 * half;
      const int* mrow = mb + (size_t)q * LL + kt + ln;
      float a0 = (mrow[0]  == 0) ? -32767.0f : s0[v] * 0.125f;
      float a1 = (mrow[16] == 0) ? -32767.0f : s1[v] * 0.125f;
      float tm = fmaxf(a0, a1);
      tm = fmaxf(tm, __shfl_xor(tm, 1));
      tm = fmaxf(tm, __shfl_xor(tm, 2));
      tm = fmaxf(tm, __shfl_xor(tm, 4));
      tm = fmaxf(tm, __shfl_xor(tm, 8));
      float mnew = fmaxf(rm[v], tm);
      float sf = __expf(rm[v] - mnew);
      float p0 = __expf(a0 - mnew);
      float p1 = __expf(a1 - mnew);
      float rs = p0 + p1;
      rs += __shfl_xor(rs, 1);
      rs += __shfl_xor(rs, 2);
      rs += __shfl_xor(rs, 4);
      rs += __shfl_xor(rs, 8);
      rl[v] = rl[v] * sf + rs;
      rm[v] = mnew;
      z0[v] *= sf; z1[v] *= sf; z2[v] *= sf; z3[v] *= sf;
      // stage probabilities to LDS (C layout -> row-major 16x32 tile)
      int m = v + 8 * half;
      lw[m * 32 + ln]      = f2bf(p0);
      lw[m * 32 + 16 + ln] = f2bf(p1);
    }
    __syncthreads();

    // ---- z += alpha(16x32) @ V(32k x 64d) : 4 WMMAs (one per 16-d tile) ----
    v16bf af = load_frag(lw, 32, lane);
    z0 = WMMA_BF16(af, load_frag(Vb + kt,                    LL, lane), z0);
    z1 = WMMA_BF16(af, load_frag(Vb + (size_t)16 * LL + kt, LL, lane), z1);
    z2 = WMMA_BF16(af, load_frag(Vb + (size_t)32 * LL + kt, LL, lane), z2);
    z3 = WMMA_BF16(af, load_frag(Vb + (size_t)48 * LL + kt, LL, lane), z3);
    __syncthreads();
  }

  // ---- normalize and store z (bf16, [B*L, H*DK] row-major) ----
#pragma unroll
  for (int v = 0; v < 8; v++) {
    int q = q0 + v + 8 * half;
    float inv = 1.0f / rl[v];
    size_t base = ((size_t)b * LL + q) * FEA + h * DKD + ln;
    Zb[base + 0]  = f2bf(z0[v] * inv);
    Zb[base + 16] = f2bf(z1[v] * inv);
    Zb[base + 32] = f2bf(z2[v] * inv);
    Zb[base + 48] = f2bf(z3[v] * inv);
  }
}

// ---------------------------------------------------------------------------
// Residual add + LayerNorm.  One block (256 thr) per row of 512.
// Writes fp32 (for next residual / final output) and optionally bf16 copy.
// ---------------------------------------------------------------------------
__global__ __launch_bounds__(256) void ln_kernel(
    const float* __restrict__ a, const float* __restrict__ b,
    const float* __restrict__ g, const float* __restrict__ be,
    float* __restrict__ outF, u16* __restrict__ outB) {
  int row = blockIdx.x, t = threadIdx.x;
  const float* pa = a + (size_t)row * FEA;
  const float* pb = b + (size_t)row * FEA;
  float v0 = pa[t] + pb[t];
  float v1 = pa[t + 256] + pb[t + 256];
  __shared__ float sh1[256], sh2[256];
  sh1[t] = v0 + v1;
  sh2[t] = v0 * v0 + v1 * v1;
  __syncthreads();
  for (int o = 128; o > 0; o >>= 1) {
    if (t < o) { sh1[t] += sh1[t + o]; sh2[t] += sh2[t + o]; }
    __syncthreads();
  }
  float mu = sh1[0] * (1.0f / FEA);
  float var = sh2[0] * (1.0f / FEA) - mu * mu;
  float inv = rsqrtf(var + 1e-5f);
  float r0 = (v0 - mu) * inv * g[t] + be[t];
  float r1 = (v1 - mu) * inv * g[t + 256] + be[t + 256];
  outF[(size_t)row * FEA + t] = r0;
  outF[(size_t)row * FEA + t + 256] = r1;
  if (outB) {
    outB[(size_t)row * FEA + t] = f2bf(r0);
    outB[(size_t)row * FEA + t + 256] = f2bf(r1);
  }
}

// ---------------------------------------------------------------------------
// Host launcher
// ---------------------------------------------------------------------------
extern "C" void kernel_launch(void* const* d_in, const int* in_sizes, int n_in,
                              void* d_out, int out_size, void* d_ws, size_t ws_size,
                              hipStream_t stream) {
  (void)in_sizes; (void)n_in; (void)out_size; (void)ws_size;

  const float* qx = (const float*)d_in[0];
  const float* kx = (const float*)d_in[1];
  const float* vx = (const float*)d_in[2];
  const int*   mk = (const int*)d_in[3];
  const float* wq = (const float*)d_in[4];
  const float* bq = (const float*)d_in[5];
  const float* wk = (const float*)d_in[6];
  const float* bk = (const float*)d_in[7];
  const float* wv = (const float*)d_in[8];
  const float* bv = (const float*)d_in[9];
  const float* wo = (const float*)d_in[10];
  const float* bo = (const float*)d_in[11];
  const float* w1 = (const float*)d_in[12];
  const float* b1 = (const float*)d_in[13];
  const float* w2 = (const float*)d_in[14];
  const float* b2 = (const float*)d_in[15];
  const float* g1 = (const float*)d_in[16];
  const float* be1 = (const float*)d_in[17];
  const float* g2 = (const float*)d_in[18];
  const float* be2 = (const float*)d_in[19];
  float* out = (float*)d_out;

  char* ws = (char*)d_ws;
  size_t off = 0;
  auto alloc = [&](size_t bytes) -> void* {
    void* p = ws + off;
    off = (off + bytes + 255) & ~(size_t)255;
    return p;
  };

  const size_t NA = (size_t)MTOT * FEA;   // 8192*512
  u16* qxb = (u16*)alloc(NA * 2);
  u16* kxb = (u16*)alloc(NA * 2);
  u16* vxb = (u16*)alloc(NA * 2);
  u16* wqT = (u16*)alloc((size_t)FEA * FEA * 2);
  u16* wkT = (u16*)alloc((size_t)FEA * FEA * 2);
  u16* wvT = (u16*)alloc((size_t)FEA * FEA * 2);
  u16* woT = (u16*)alloc((size_t)FEA * FEA * 2);
  u16* w1T = (u16*)alloc((size_t)FF * FEA * 2);
  u16* w2T = (u16*)alloc((size_t)FEA * FF * 2);
  u16* Qp  = (u16*)alloc(NA * 2);
  u16* Kp  = (u16*)alloc(NA * 2);
  u16* Vt  = (u16*)alloc(NA * 2);
  u16* Zb  = (u16*)alloc(NA * 2);
  float* S1 = (float*)alloc(NA * 4);
  float* ZN = (float*)alloc(NA * 4);
  u16* ZNb = (u16*)alloc(NA * 2);
  u16* Hb  = (u16*)alloc((size_t)MTOT * FF * 2);
  float* S2 = (float*)alloc(NA * 4);

  // 1) convert activations to bf16
  {
    int n = (int)NA, blk = 256, grd = (n + blk - 1) / blk;
    cvt_bf16_kernel<<<grd, blk, 0, stream>>>(qx, qxb, n);
    cvt_bf16_kernel<<<grd, blk, 0, stream>>>(kx, kxb, n);
    cvt_bf16_kernel<<<grd, blk, 0, stream>>>(vx, vxb, n);
  }
  // 2) transpose+convert weights: [K,N] fp32 -> [N,K] bf16
  {
    int n = FEA * FEA, blk = 256, grd = (n + blk - 1) / blk;
    cvt_tr_bf16_kernel<<<grd, blk, 0, stream>>>(wq, wqT, FEA, FEA);
    cvt_tr_bf16_kernel<<<grd, blk, 0, stream>>>(wk, wkT, FEA, FEA);
    cvt_tr_bf16_kernel<<<grd, blk, 0, stream>>>(wv, wvT, FEA, FEA);
    cvt_tr_bf16_kernel<<<grd, blk, 0, stream>>>(wo, woT, FEA, FEA);
    int n2 = FEA * FF, grd2 = (n2 + blk - 1) / blk;
    cvt_tr_bf16_kernel<<<grd2, blk, 0, stream>>>(w1, w1T, FEA, FF);
    cvt_tr_bf16_kernel<<<grd2, blk, 0, stream>>>(w2, w2T, FF, FEA);
  }
  // 3) Q/K/V projections (WMMA GEMMs)
  {
    dim3 grd(MTOT / 64, FEA / 128);
    gemm_bf16_kernel<1><<<grd, 256, 0, stream>>>(qxb, wqT, bq, nullptr, Qp, MTOT, FEA, FEA);
    gemm_bf16_kernel<1><<<grd, 256, 0, stream>>>(kxb, wkT, bk, nullptr, Kp, MTOT, FEA, FEA);
    gemm_bf16_kernel<2><<<grd, 256, 0, stream>>>(vxb, wvT, bv, nullptr, Vt, MTOT, FEA, FEA);
  }
  // 4) attention (flash, online softmax, WMMA)
  attn_kernel<<<(BB * HH * (LL / 16)) / 4, 128, 0, stream>>>(Qp, Kp, Vt, mk, Zb);
  // 5) output projection -> f32
  {
    dim3 grd(MTOT / 64, FEA / 128);
    gemm_bf16_kernel<0><<<grd, 256, 0, stream>>>(Zb, woT, bo, S1, nullptr, MTOT, FEA, FEA);
  }
  // 6) LN1: zn = LN(vx + attn_out)
  ln_kernel<<<MTOT, 256, 0, stream>>>(vx, S1, g1, be1, ZN, ZNb);
  // 7) FFN1 with exact GELU epilogue
  {
    dim3 grd(MTOT / 64, FF / 128);
    gemm_bf16_kernel<3><<<grd, 256, 0, stream>>>(ZNb, w1T, b1, nullptr, Hb, MTOT, FF, FEA);
  }
  // 8) FFN2 -> f32
  {
    dim3 grd(MTOT / 64, FEA / 128);
    gemm_bf16_kernel<0><<<grd, 256, 0, stream>>>(Hb, w2T, b2, S2, nullptr, MTOT, FEA, FF);
  }
  // 9) LN2 -> final output
  ln_kernel<<<MTOT, 256, 0, stream>>>(ZN, S2, g2, be2, out, nullptr);
}
